// ScaleDotProductAttention_48687749267512
// MI455X (gfx1250) — matrix operand order repaired and checked by reading, not compiled, not run on hardware
//
#include <hip/hip_runtime.h>

#define NB 16
#define LL 2048
#define DM 1024
#define DK 128
#define DV 128

#define KTS  136  // Kt row stride in ushorts (272B, 16B-aligned, bank-spread)
#define WTS  40   // proj Wt row stride in ushorts (80B)
#define VTS2 72   // Vt/Pb row stride in ushorts (144B, 16B-aligned, bank-spread)

typedef __attribute__((ext_vector_type(16))) __bf16 v16bf;
typedef __attribute__((ext_vector_type(8)))  float  v8f;

union BF16x16 { v16bf v; __bf16 h[16]; unsigned short s[16]; uint4 q[2]; };
union BF16x8  { uint4 q;  __bf16 h[8]; };

#if __has_builtin(__builtin_amdgcn_exp2f)
#define EXP2F(x) __builtin_amdgcn_exp2f(x)   // bare v_exp_f32; underflow->0 is fine here
#else
#define EXP2F(x) exp2f(x)
#endif

__device__ __forceinline__ unsigned short bf_bits(float f) {
  union { __bf16 h; unsigned short u; } c; c.h = (__bf16)f; return c.u;
}

// ---- CDNA5 async global->LDS copy (CDNA5 ISA, GV mode) -------
__device__ __forceinline__ void async_ld_b128(unsigned lds_addr, const void* gaddr) {
  asm volatile("global_load_async_to_lds_b128 %0, %1, off"
               :: "v"(lds_addr), "v"(gaddr) : "memory");
}
__device__ __forceinline__ void wait_async0() {
  asm volatile("s_wait_asynccnt 0x0" ::: "memory");
}
__device__ __forceinline__ unsigned lds_off(const void* p) {
  return (unsigned)(uintptr_t)p;   // LDS aperture: addr[31:0] is the LDS byte offset
}

// ---------------------------------------------------------------------------
// Projection: Y = X[32768x1024] @ W[1024x128] -> bf16.
// TRANS=false: Y[row][col] row-major.  TRANS=true: Y[n][col][l] (V transposed).
// ---------------------------------------------------------------------------
template <bool TRANS>
__global__ __launch_bounds__(128)
void proj_kernel(const float* __restrict__ X, const float* __restrict__ W,
                 unsigned short* __restrict__ Yp) {
  __shared__ unsigned short Wt[128 * WTS];       // [col][k] chunk, padded rows

  const int tid  = threadIdx.x;
  const int lane = tid & 31;
  const int wave = tid >> 5;
  const int lo   = lane & 15;
  const int hi   = lane >> 4;
  const int row0 = blockIdx.x * 64 + wave * 16;

  v8f zero = {};
  v8f acc[8];
#pragma unroll
  for (int t = 0; t < 8; ++t) acc[t] = zero;

  for (int d0 = 0; d0 < DM; d0 += 32) {
    __syncthreads();
    {   // stage W[d0..d0+31][0..127] transposed -> Wt[col][k], bf16
      int k  = tid >> 2;
      int c0 = (tid & 3) * 32;
      const float* wrow = W + (size_t)(d0 + k) * DK + c0;
#pragma unroll
      for (int cc = 0; cc < 8; ++cc) {
        float4 f = *(const float4*)(wrow + cc * 4);
        Wt[(c0 + cc * 4 + 0) * WTS + k] = bf_bits(f.x);
        Wt[(c0 + cc * 4 + 1) * WTS + k] = bf_bits(f.y);
        Wt[(c0 + cc * 4 + 2) * WTS + k] = bf_bits(f.z);
        Wt[(c0 + cc * 4 + 3) * WTS + k] = bf_bits(f.w);
      }
    }
    __syncthreads();

    // A fragment: 16 rows of X, K chunk of 32 (native f32->bf16 casts)
    BF16x16 a;
    {
      const float* xr = X + (size_t)(row0 + lo) * DM + d0 + hi * 8;
      float4 f0 = *(const float4*)(xr);
      float4 f1 = *(const float4*)(xr + 4);
      float4 f2 = *(const float4*)(xr + 16);
      float4 f3 = *(const float4*)(xr + 20);
      a.h[0] = (__bf16)f0.x;  a.h[1] = (__bf16)f0.y;
      a.h[2] = (__bf16)f0.z;  a.h[3] = (__bf16)f0.w;
      a.h[4] = (__bf16)f1.x;  a.h[5] = (__bf16)f1.y;
      a.h[6] = (__bf16)f1.z;  a.h[7] = (__bf16)f1.w;
      a.h[8]  = (__bf16)f2.x; a.h[9]  = (__bf16)f2.y;
      a.h[10] = (__bf16)f2.z; a.h[11] = (__bf16)f2.w;
      a.h[12] = (__bf16)f3.x; a.h[13] = (__bf16)f3.y;
      a.h[14] = (__bf16)f3.z; a.h[15] = (__bf16)f3.w;
    }

    // preload all B fragments, then the WMMA chain
    BF16x16 b[8];
#pragma unroll
    for (int t = 0; t < 8; ++t) {
      const unsigned short* src = &Wt[(t * 16 + lo) * WTS + hi * 8];
      b[t].q[0] = *(const uint4*)(src);
      b[t].q[1] = *(const uint4*)(src + 16);
    }
#pragma unroll
    for (int t = 0; t < 8; ++t)
      acc[t] = __builtin_amdgcn_wmma_f32_16x16x32_bf16(
          false, a.v, false, b[t].v, (short)0, acc[t], false, false);
  }

  if (TRANS) {
    // V path: store transposed  Yp[n][col][l], 8 consecutive keys per store
    int n  = row0 / LL;
    int l0 = (row0 % LL) + hi * 8;
#pragma unroll
    for (int t = 0; t < 8; ++t) {
      BF16x8 pk;
#pragma unroll
      for (int r = 0; r < 8; ++r) pk.h[r] = (__bf16)acc[t][r];
      *(uint4*)(Yp + ((size_t)n * DK + t * 16 + lo) * LL + l0) = pk.q;
    }
  } else {
#pragma unroll
    for (int t = 0; t < 8; ++t)
#pragma unroll
      for (int r = 0; r < 8; ++r)
        Yp[(size_t)(row0 + r + hi * 8) * DK + t * 16 + lo] = bf_bits(acc[t][r]);
  }
}

// ---------------------------------------------------------------------------
// Flash attention: grid (L/64, N), 128 threads = 4 waves, key block = 64.
// K tile + pre-transposed V tile staged via async global->LDS b128.
// Online softmax in log2 domain; row-sums computed by WMMA against ones.
// ---------------------------------------------------------------------------
__global__ __launch_bounds__(128)
void attn_kernel(const unsigned short* __restrict__ Qp,
                 const unsigned short* __restrict__ Kp,
                 const unsigned short* __restrict__ Vpt,   // [n][col][key]
                 const unsigned char* __restrict__ maskp,
                 float* __restrict__ out) {
  __shared__ unsigned short Kt[64 * KTS];        // [key][d], padded
  __shared__ unsigned short Vt[128 * VTS2];      // [v-col][key], padded
  __shared__ unsigned short Pb[4][16 * VTS2];    // per-wave P scratch [row][k]

  const int tid  = threadIdx.x;
  const int lane = tid & 31;
  const int wave = tid >> 5;
  const int lo   = lane & 15;
  const int hi   = lane >> 4;
  const int n    = blockIdx.y;
  const int qb   = blockIdx.x * 64;
  const int q0   = qb + wave * 16;

  // Q fragments for this wave: 4 chunks of K=32 (16x128 bf16)
  BF16x16 qf[4];
  {
    const unsigned short* qr = Qp + ((size_t)n * LL + q0 + lo) * DK + hi * 8;
#pragma unroll
    for (int c = 0; c < 4; ++c) {
      qf[c].q[0] = *(const uint4*)(qr + c * 32);
      qf[c].q[1] = *(const uint4*)(qr + c * 32 + 16);
    }
  }

  // all-ones B fragment: P x ones => per-row sums broadcast to every lane
  BF16x16 onesb;
#pragma unroll
  for (int i = 0; i < 16; ++i) onesb.h[i] = (__bf16)1.0f;

  v8f zero = {};
  v8f o[8];
#pragma unroll
  for (int t = 0; t < 8; ++t) o[t] = zero;
  float mrun[8], lrun[8];
#pragma unroll
  for (int r = 0; r < 8; ++r) { mrun[r] = -1e30f; lrun[r] = 0.f; }

  const int   num_kb = blockIdx.x + 1;            // causal: keys up to qb+63
  const float scl2   = 0.088388347648318447f * 1.4426950408889634f; // 1/sqrt(128)*log2(e)

  for (int kb = 0; kb < num_kb; ++kb) {
    const int k0 = kb * 64;
    __syncthreads();
    {   // async stage: 64 K rows + 128 pre-transposed V rows (128B/thread each)
      int krow = tid >> 1;
      int c0   = (tid & 1) * 64;
      const unsigned short* ksrc = Kp + ((size_t)n * LL + k0 + krow) * DK + c0;
      unsigned kdst = lds_off(&Kt[krow * KTS + c0]);
#pragma unroll
      for (int j = 0; j < 8; ++j)
        async_ld_b128(kdst + j * 16, ksrc + j * 8);
      const unsigned short* vsrc = Vpt + ((size_t)n * DK + tid) * LL + k0;
      unsigned vdst = lds_off(&Vt[tid * VTS2]);
#pragma unroll
      for (int j = 0; j < 8; ++j)
        async_ld_b128(vdst + j * 16, vsrc + j * 8);
    }
    wait_async0();
    __syncthreads();

    // S = Q . K^T for 64 keys: four 16x16 tiles, K-dim 128 in 4 chunks each
    v8f s[4];
#pragma unroll
    for (int j = 0; j < 4; ++j) {
      BF16x16 kbf[4];
#pragma unroll
      for (int c = 0; c < 4; ++c) {
        const unsigned short* p = &Kt[(j * 16 + lo) * KTS + c * 32 + hi * 8];
        kbf[c].q[0] = *(const uint4*)p;
        kbf[c].q[1] = *(const uint4*)(p + 16);
      }
      s[j] = zero;
#pragma unroll
      for (int c = 0; c < 4; ++c)
        s[j] = __builtin_amdgcn_wmma_f32_16x16x32_bf16(
            false, qf[c].v, false, kbf[c].v, (short)0, s[j], false, false);
    }

    // scale(+log2e) + causal/padding mask + row max (local over 4 tiles first)
    float sv[4][8], tmax[8];
#pragma unroll
    for (int r = 0; r < 8; ++r) {
      int q = q0 + r + hi * 8;
      size_t mbase = ((size_t)n * LL + q) * LL;
      float tm = -1e30f;
#pragma unroll
      for (int j = 0; j < 4; ++j) {
        int kk = k0 + j * 16 + lo;
        float a = s[j][r] * scl2;
        if (kk > q || maskp[mbase + kk]) a = -1e30f;
        sv[j][r] = a;
        tm = fmaxf(tm, a);
      }
#pragma unroll
      for (int off = 8; off >= 1; off >>= 1)
        tm = fmaxf(tm, __shfl_xor(tm, off, 32));
      tmax[r] = tm;
    }

    // online softmax update (log2 domain); write P (bf16) into per-wave LDS
    float alpha[8];
#pragma unroll
    for (int r = 0; r < 8; ++r) {
      float mnew = fmaxf(mrun[r], tmax[r]);
      alpha[r] = EXP2F(mrun[r] - mnew);
      mrun[r] = mnew;
#pragma unroll
      for (int t = 0; t < 8; ++t) o[t][r] *= alpha[r];
      int prow = r + hi * 8;
#pragma unroll
      for (int j = 0; j < 4; ++j)
        Pb[wave][prow * VTS2 + j * 16 + lo] = bf_bits(EXP2F(sv[j][r] - mnew));
    }

    // P A-fragments (16x64 in 2 chunks) from per-wave LDS scratch
    BF16x16 pa[2];
#pragma unroll
    for (int c = 0; c < 2; ++c) {
      const unsigned short* pr = &Pb[wave][lo * VTS2 + c * 32 + hi * 8];
      pa[c].q[0] = *(const uint4*)pr;
      pa[c].q[1] = *(const uint4*)(pr + 16);
    }

    // row sums via WMMA: ls[r] = sum_k P[row_r, k], identical in every lane
    v8f ls = zero;
#pragma unroll
    for (int c = 0; c < 2; ++c)
      ls = __builtin_amdgcn_wmma_f32_16x16x32_bf16(
          false, pa[c].v, false, onesb.v, (short)0, ls, false, false);
#pragma unroll
    for (int r = 0; r < 8; ++r)
      lrun[r] = lrun[r] * alpha[r] + ls[r];

    // O += P (16x64) . V (64x128): 2 A-chunks x 8 N-tiles
#pragma unroll
    for (int t = 0; t < 8; ++t) {
#pragma unroll
      for (int c = 0; c < 2; ++c) {
        BF16x16 bv;
        const unsigned short* vp_ = &Vt[(t * 16 + lo) * VTS2 + c * 32 + hi * 8];
        bv.q[0] = *(const uint4*)vp_;
        bv.q[1] = *(const uint4*)(vp_ + 16);
        o[t] = __builtin_amdgcn_wmma_f32_16x16x32_bf16(
            false, pa[c].v, false, bv.v, (short)0, o[t], false, false);
      }
    }
  }

  // normalize and store fp32 output
#pragma unroll
  for (int r = 0; r < 8; ++r) {
    float inv = lrun[r] > 0.f ? 1.0f / lrun[r] : 0.f;
    int q = q0 + r + hi * 8;
    float* orow = out + ((size_t)n * LL + q) * DV;
#pragma unroll
    for (int t = 0; t < 8; ++t)
      orow[t * 16 + lo] = o[t][r] * inv;
  }
}

extern "C" void kernel_launch(void* const* d_in, const int* in_sizes, int n_in,
                              void* d_out, int out_size, void* d_ws, size_t ws_size,
                              hipStream_t stream) {
  const float* Q  = (const float*)d_in[0];
  const float* K  = (const float*)d_in[1];
  const float* V  = (const float*)d_in[2];
  const unsigned char* mask = (const unsigned char*)d_in[3];
  const float* WQ = (const float*)d_in[4];
  const float* WK = (const float*)d_in[5];
  const float* WV = (const float*)d_in[6];
  float* out = (float*)d_out;

  unsigned short* Qp  = (unsigned short*)d_ws;                 // 8 MB bf16
  unsigned short* Kp  = Qp + (size_t)NB * LL * DK;             // 8 MB bf16
  unsigned short* Vpt = Kp + (size_t)NB * LL * DK;             // 8 MB bf16, transposed

  dim3 pgrid(NB * LL / 64);
  proj_kernel<false><<<pgrid, 128, 0, stream>>>(Q, WQ, Qp);
  proj_kernel<false><<<pgrid, 128, 0, stream>>>(K, WK, Kp);
  proj_kernel<true ><<<pgrid, 128, 0, stream>>>(V, WV, Vpt);

  dim3 agrid(LL / 64, NB);
  attn_kernel<<<agrid, 128, 0, stream>>>(Qp, Kp, Vpt, mask, out);
}
